// ABL_NODE_62560493634184
// MI455X (gfx1250) — matrix operand order repaired and checked by reading, compiled, not run on hardware
//
#include <hip/hip_runtime.h>

#define NUM_GRAPHS 512
#define GRAPH_SIZE 48
#define MAX_NN 64
#define DDIM 128
#define MSGD 256
#define TOTAL_NODES (NUM_GRAPHS * GRAPH_SIZE)   /* 24576 */
#define TOTAL_EDGES (TOTAL_NODES * 8)           /* 196608 */
#define PROP_STEPS 5

typedef __attribute__((ext_vector_type(16))) __bf16 v16bf;
typedef __attribute__((ext_vector_type(8)))  __bf16 v8bf;
typedef __attribute__((ext_vector_type(8)))  float  v8f;

__device__ __forceinline__ __bf16 f2bf(float f) {
  unsigned u = __builtin_bit_cast(unsigned, f);
  unsigned r = u + 0x7FFFu + ((u >> 16) & 1u);
  unsigned short hh = (unsigned short)(r >> 16);
  return __builtin_bit_cast(__bf16, hh);
}
__device__ __forceinline__ float bf2f(__bf16 b) {
  unsigned short hh = __builtin_bit_cast(unsigned short, b);
  unsigned u = ((unsigned)hh) << 16;
  return __builtin_bit_cast(float, u);
}
__device__ __forceinline__ v8f v8f_zero() {
  v8f z = {0.f, 0.f, 0.f, 0.f, 0.f, 0.f, 0.f, 0.f};
  return z;
}

// A fragment (16x32 bf16, MxK): lane L holds row M=L%16; kb=(L/16)*8.
// j<8 -> K = kb+j ; j>=8 -> K = 16+kb+(j-8). Two contiguous 16B loads per lane.
__device__ __forceinline__ v16bf load_a16(const __bf16* rowp, int kb) {
  v8bf lo = *(const v8bf*)(rowp + kb);
  v8bf hi = *(const v8bf*)(rowp + kb + 16);
  return __builtin_shufflevector(lo, hi, 0, 1, 2, 3, 4, 5, 6, 7,
                                 8, 9, 10, 11, 12, 13, 14, 15);
}
// B fragment: pre-packed contiguous 16 bf16 per lane.
__device__ __forceinline__ v16bf load_b16(const __bf16* p) {
  v8bf lo = *(const v8bf*)(p);
  v8bf hi = *(const v8bf*)(p + 8);
  return __builtin_shufflevector(lo, hi, 0, 1, 2, 3, 4, 5, 6, 7,
                                 8, 9, 10, 11, 12, 13, 14, 15);
}

// ---------- weight packing: fp32 row-major (K x N) -> bf16 B-fragment layout
// packed[((kc*(N/16)+nt)*32 + lane)*16 + j] = W[(kc*32 + (lane/16)*8 + (j%8) + (j/8)*16)*N + nt*16 + lane%16]
__global__ void pack_w_kernel(const float* __restrict__ W, __bf16* __restrict__ P,
                              int K, int N) {
  int tid = blockIdx.x * blockDim.x + threadIdx.x;
  if (tid >= K * N) return;
  int j = tid & 15;
  int lane = (tid >> 4) & 31;
  int rest = tid >> 9;
  int NT = N >> 4;
  int nt = rest % NT;
  int kc = rest / NT;
  int n = nt * 16 + (lane & 15);
  int k = kc * 32 + (lane >> 4) * 8 + (j & 7) + ((j >> 3) << 4);
  P[tid] = f2bf(W[(size_t)k * N + n]);
}

// ---------- encoders (tiny K; scalar fp32) ----------
__global__ void enc_node_kernel(const float* __restrict__ nf, const float* __restrict__ W,
                                const float* __restrict__ b, float* __restrict__ h,
                                __bf16* __restrict__ hbf) {
  int tid = blockIdx.x * blockDim.x + threadIdx.x;  // TOTAL_NODES*128
  int col = tid & 127;
  int node = tid >> 7;
  const float* x = nf + (size_t)node * 32;
  float acc = b[col];
#pragma unroll
  for (int k = 0; k < 32; k++) acc += x[k] * W[k * 128 + col];
  h[tid] = acc;
  hbf[tid] = f2bf(acc);
}

__global__ void enc_edge_kernel(const float* __restrict__ ef, const float* __restrict__ W,
                                const float* __restrict__ b, __bf16* __restrict__ ebf) {
  int tid = blockIdx.x * blockDim.x + threadIdx.x;  // TOTAL_EDGES*128
  int col = tid & 127;
  int ed = tid >> 7;
  const float* x = ef + (size_t)ed * 16;
  float acc = b[col];
#pragma unroll
  for (int k = 0; k < 16; k++) acc += x[k] * W[k * 128 + col];
  ebf[tid] = f2bf(acc);
}

// ---------- message kernel: per 16-edge tile, fwd+rev 384->256->256 MLPs,
// scatter-add into agg (fp32 atomics). Block = 128 thr = 4 waves:
// wave = dir*2 + half, dir: 0=fwd(m),1=rev(rm); half: N range [half*128, half*128+128)
__global__ void __launch_bounds__(128) msg_kernel(
    const __bf16* __restrict__ hbf, const __bf16* __restrict__ ebf,
    const int* __restrict__ from_idx, const int* __restrict__ to_idx,
    const __bf16* __restrict__ pW1m, const __bf16* __restrict__ pW2m,
    const __bf16* __restrict__ pW1r, const __bf16* __restrict__ pW2r,
    const float* __restrict__ b1m, const float* __restrict__ b2m,
    const float* __restrict__ b1r, const float* __restrict__ b2r,
    float* __restrict__ agg) {
  __shared__ int sfrom[16], sto[16];
  __shared__ __attribute__((aligned(16))) __bf16 hid[2][16][MSGD];  // 16 KB
  int tile = blockIdx.x;
  int t = threadIdx.x;
  if (t < 16) {
    sfrom[t] = from_idx[tile * 16 + t];
    sto[t] = to_idx[tile * 16 + t];
  }
  __syncthreads();
  int wave = t >> 5;
  int lane = t & 31;
  int dir = wave >> 1;
  int hfl = wave & 1;
  int m = lane & 15;
  int kb = (lane >> 4) * 8;
  int edge = tile * 16 + m;
  const __bf16* pf = hbf + (size_t)sfrom[m] * DDIM;
  const __bf16* pt = hbf + (size_t)sto[m] * DDIM;
  const __bf16* pA0 = dir ? pt : pf;
  const __bf16* pA1 = dir ? pf : pt;
  const __bf16* pE = ebf + (size_t)edge * DDIM;
  const __bf16* W1 = dir ? pW1r : pW1m;
  const __bf16* W2 = dir ? pW2r : pW2m;
  const float* bb1 = dir ? b1r : b1m;
  const float* bb2 = dir ? b2r : b2m;

  v8f acc[8];
#pragma unroll
  for (int i = 0; i < 8; i++) acc[i] = v8f_zero();

#pragma unroll
  for (int kc = 0; kc < 12; kc++) {
    const __bf16* src = (kc < 4)   ? (pA0 + kc * 32)
                        : (kc < 8) ? (pA1 + (kc - 4) * 32)
                                   : (pE + (kc - 8) * 32);
    v16bf a = load_a16(src, kb);
#pragma unroll
    for (int nt = 0; nt < 8; nt++) {
      int gnt = hfl * 8 + nt;
      v16bf bv = load_b16(W1 + ((size_t)(kc * 16 + gnt) * 32 + lane) * 16);
      acc[nt] = __builtin_amdgcn_wmma_f32_16x16x32_bf16(false, a, false, bv,
                                                        (short)0, acc[nt], false, false);
    }
  }
  // bias + relu -> LDS (bf16)
#pragma unroll
  for (int nt = 0; nt < 8; nt++) {
    int col = (hfl * 8 + nt) * 16 + m;
    float bias = bb1[col];
#pragma unroll
    for (int v = 0; v < 8; v++) {
      int row = v + (lane >> 4) * 8;
      float x = acc[nt][v] + bias;
      hid[dir][row][col] = f2bf(x > 0.f ? x : 0.f);
    }
  }
  __syncthreads();

  v8f acc2[8];
#pragma unroll
  for (int i = 0; i < 8; i++) acc2[i] = v8f_zero();
#pragma unroll
  for (int kc = 0; kc < 8; kc++) {
    v16bf a = load_a16(&hid[dir][m][kc * 32], kb);
#pragma unroll
    for (int nt = 0; nt < 8; nt++) {
      int gnt = hfl * 8 + nt;
      v16bf bv = load_b16(W2 + ((size_t)(kc * 16 + gnt) * 32 + lane) * 16);
      acc2[nt] = __builtin_amdgcn_wmma_f32_16x16x32_bf16(false, a, false, bv,
                                                         (short)0, acc2[nt], false, false);
    }
  }
  // bias + scatter-add (m -> to_idx, rm -> from_idx)
#pragma unroll
  for (int nt = 0; nt < 8; nt++) {
    int col = (hfl * 8 + nt) * 16 + m;
    float bias = bb2[col];
#pragma unroll
    for (int v = 0; v < 8; v++) {
      int row = v + (lane >> 4) * 8;
      int dest = dir ? sfrom[row] : sto[row];
      atomicAdd(agg + (size_t)dest * MSGD + col, acc2[nt][v] + bias);
    }
  }
}

// ---------- update kernel: h += mlp2([h, agg]); 16-node tiles, 4 waves
__global__ void __launch_bounds__(128) upd_kernel(
    const float* __restrict__ aggm,
    const __bf16* __restrict__ pU1, const __bf16* __restrict__ pU2,
    const float* __restrict__ ub1, const float* __restrict__ ub2,
    float* __restrict__ h, __bf16* __restrict__ hbf) {
  __shared__ __attribute__((aligned(16))) __bf16 hid[16][MSGD];  // 8 KB
  int tile = blockIdx.x;
  int t = threadIdx.x;
  int wave = t >> 5;
  int lane = t & 31;
  int m = lane & 15;
  int kb = (lane >> 4) * 8;
  int row0 = tile * 16;
  const __bf16* ph = hbf + (size_t)(row0 + m) * DDIM;
  const float* pa = aggm + (size_t)(row0 + m) * MSGD;

  v8f acc[4];
#pragma unroll
  for (int i = 0; i < 4; i++) acc[i] = v8f_zero();
#pragma unroll
  for (int kc = 0; kc < 12; kc++) {
    v16bf a;
    if (kc < 4) {
      a = load_a16(ph + kc * 32, kb);
    } else {
      const float* s = pa + (kc - 4) * 32;
      v16bf tmp;
#pragma unroll
      for (int j = 0; j < 8; j++) tmp[j] = f2bf(s[kb + j]);
#pragma unroll
      for (int j = 0; j < 8; j++) tmp[8 + j] = f2bf(s[16 + kb + j]);
      a = tmp;
    }
#pragma unroll
    for (int nt = 0; nt < 4; nt++) {
      int gnt = wave * 4 + nt;
      v16bf bv = load_b16(pU1 + ((size_t)(kc * 16 + gnt) * 32 + lane) * 16);
      acc[nt] = __builtin_amdgcn_wmma_f32_16x16x32_bf16(false, a, false, bv,
                                                        (short)0, acc[nt], false, false);
    }
  }
#pragma unroll
  for (int nt = 0; nt < 4; nt++) {
    int col = (wave * 4 + nt) * 16 + m;
    float bias = ub1[col];
#pragma unroll
    for (int v = 0; v < 8; v++) {
      int row = v + (lane >> 4) * 8;
      float x = acc[nt][v] + bias;
      hid[row][col] = f2bf(x > 0.f ? x : 0.f);
    }
  }
  __syncthreads();

  v8f acc2[2];
  acc2[0] = v8f_zero();
  acc2[1] = v8f_zero();
#pragma unroll
  for (int kc = 0; kc < 8; kc++) {
    v16bf a = load_a16(&hid[m][kc * 32], kb);
#pragma unroll
    for (int nt = 0; nt < 2; nt++) {
      int gnt = wave * 2 + nt;
      v16bf bv = load_b16(pU2 + ((size_t)(kc * 8 + gnt) * 32 + lane) * 16);
      acc2[nt] = __builtin_amdgcn_wmma_f32_16x16x32_bf16(false, a, false, bv,
                                                         (short)0, acc2[nt], false, false);
    }
  }
#pragma unroll
  for (int nt = 0; nt < 2; nt++) {
    int col = (wave * 2 + nt) * 16 + m;
    float bias = ub2[col];
#pragma unroll
    for (int v = 0; v < 8; v++) {
      int row = v + (lane >> 4) * 8;
      size_t idx = (size_t)(row0 + row) * DDIM + col;
      float nh = h[idx] + acc2[nt][v] + bias;
      h[idx] = nh;
      hbf[idx] = f2bf(nh);
    }
  }
}

// ---------- sink MLP per graph: t[g] = relu(pad(h_g)@W1 + b1)@W2 + b2  (64x64)
__global__ void __launch_bounds__(256) sink_t_kernel(
    const float* __restrict__ h, const float* __restrict__ W1, const float* __restrict__ b1,
    const float* __restrict__ W2, const float* __restrict__ b2, float* __restrict__ tbuf) {
  __shared__ float x[MAX_NN][DDIM];    // 32 KB
  __shared__ float hid[MAX_NN][64];    // 16 KB
  int g = blockIdx.x;
  int t = threadIdx.x;
  for (int i = t; i < MAX_NN * DDIM; i += 256) {
    int r = i >> 7, d = i & 127;
    x[r][d] = (r < GRAPH_SIZE) ? h[((size_t)g * GRAPH_SIZE + r) * DDIM + d] : 0.f;
  }
  __syncthreads();
  for (int i = t; i < MAX_NN * 64; i += 256) {
    int r = i >> 6, j = i & 63;
    float acc = b1[j];
    for (int d = 0; d < DDIM; d++) acc += x[r][d] * W1[d * 64 + j];
    hid[r][j] = acc > 0.f ? acc : 0.f;
  }
  __syncthreads();
  for (int i = t; i < MAX_NN * 64; i += 256) {
    int r = i >> 6, j = i & 63;
    float acc = b2[j];
    for (int k = 0; k < 64; k++) acc += hid[r][k] * W2[k * 64 + j];
    tbuf[(size_t)g * 4096 + i] = acc;
  }
}

// ---------- per-pair Sinkhorn + final reduction ----------
__global__ void __launch_bounds__(256) pair_kernel(
    const float* __restrict__ h, const float* __restrict__ tbuf, float* __restrict__ out) {
  __shared__ __attribute__((aligned(16))) float smem[12544];  // ~49 KB
  float* tq = smem;           // 4096
  float* tc = smem + 4096;    // 4096
  float* la = smem + 8192;    // 4096
  float* red = smem + 12288;  // 256
  __bf16* qb = (__bf16*)smem;           // 64x128 (aliases tq/tc after Sinkhorn)
  __bf16* cb = (__bf16*)(smem + 4096);  // 64x128
  int p = blockIdx.x;
  int t = threadIdx.x;
  for (int i = t; i < 4096; i += 256) {
    tq[i] = tbuf[(size_t)(2 * p) * 4096 + i];
    tc[i] = tbuf[(size_t)(2 * p + 1) * 4096 + i];
  }
  __syncthreads();
  for (int i = t; i < 4096; i += 256) {
    int r = i >> 6, j = i & 63;
    float s = 0.f;
    for (int k = 0; k < 64; k++) s += fabsf(tq[r * 64 + k] - tc[j * 64 + k]);
    la[i] = -s * 10.0f;  // -cost / 0.1
  }
  __syncthreads();
  for (int it = 0; it < 20; it++) {
    if (t < 64) {  // LSE over axis=2 (columns j) for row t
      float mx = -1e30f;
      for (int j = 0; j < 64; j++) mx = fmaxf(mx, la[t * 64 + j]);
      float s = 0.f;
      for (int j = 0; j < 64; j++) s += __expf(la[t * 64 + j] - mx);
      float l = mx + __logf(s);
      for (int j = 0; j < 64; j++) la[t * 64 + j] -= l;
    }
    __syncthreads();
    if (t < 64) {  // LSE over axis=1 (rows i) for column t
      float mx = -1e30f;
      for (int r = 0; r < 64; r++) mx = fmaxf(mx, la[r * 64 + t]);
      float s = 0.f;
      for (int r = 0; r < 64; r++) s += __expf(la[r * 64 + t] - mx);
      float l = mx + __logf(s);
      for (int r = 0; r < 64; r++) la[r * 64 + t] -= l;
    }
    __syncthreads();
  }
  // stage q (graph 2p) and c (graph 2p+1) as bf16, reusing tq/tc space
  for (int i = t; i < MAX_NN * DDIM; i += 256) {
    int r = i >> 7, d = i & 127;
    qb[i] = (r < GRAPH_SIZE) ? f2bf(h[((size_t)(2 * p) * GRAPH_SIZE + r) * DDIM + d]) : f2bf(0.f);
    cb[i] = (r < GRAPH_SIZE) ? f2bf(h[((size_t)(2 * p + 1) * GRAPH_SIZE + r) * DDIM + d]) : f2bf(0.f);
  }
  __syncthreads();
  float accv = 0.f;
  for (int i = t; i < 4096; i += 256) {
    int r = i >> 6, j = i & 63;
    const __bf16* qr = qb + r * DDIM;
    const __bf16* cr = cb + j * DDIM;
    float pd = 0.f;
    for (int d = 0; d < DDIM; d++) pd += fabsf(bf2f(qr[d]) - bf2f(cr[d]));
    accv += __expf(la[i]) * pd;
  }
  red[t] = accv;
  __syncthreads();
  for (int s = 128; s > 0; s >>= 1) {
    if (t < s) red[t] += red[t + s];
    __syncthreads();
  }
  if (t == 0) out[p] = red[0];
}

extern "C" void kernel_launch(void* const* d_in, const int* in_sizes, int n_in,
                              void* d_out, int out_size, void* d_ws, size_t ws_size,
                              hipStream_t stream) {
  (void)in_sizes; (void)n_in; (void)out_size; (void)ws_size;
  const float* node_features = (const float*)d_in[0];
  const float* edge_features = (const float*)d_in[1];
  const int* from_idx = (const int*)d_in[2];
  const int* to_idx = (const int*)d_in[3];
  const float* enc_node_W = (const float*)d_in[4];
  const float* enc_node_b = (const float*)d_in[5];
  const float* enc_edge_W = (const float*)d_in[6];
  const float* enc_edge_b = (const float*)d_in[7];
  const float* msg_W1 = (const float*)d_in[8];
  const float* msg_b1 = (const float*)d_in[9];
  const float* msg_W2 = (const float*)d_in[10];
  const float* msg_b2 = (const float*)d_in[11];
  const float* rmsg_W1 = (const float*)d_in[12];
  const float* rmsg_b1 = (const float*)d_in[13];
  const float* rmsg_W2 = (const float*)d_in[14];
  const float* rmsg_b2 = (const float*)d_in[15];
  const float* upd_W1 = (const float*)d_in[16];
  const float* upd_b1 = (const float*)d_in[17];
  const float* upd_W2 = (const float*)d_in[18];
  const float* upd_b2 = (const float*)d_in[19];
  const float* sink_W1 = (const float*)d_in[20];
  const float* sink_b1 = (const float*)d_in[21];
  const float* sink_W2 = (const float*)d_in[22];
  const float* sink_b2 = (const float*)d_in[23];
  float* out = (float*)d_out;

  char* ws = (char*)d_ws;
  size_t off = 0;
  auto alloc = [&](size_t bytes) -> char* {
    char* p = ws + off;
    off += (bytes + 255) & ~(size_t)255;
    return p;
  };
  float* h = (float*)alloc((size_t)TOTAL_NODES * DDIM * 4);
  __bf16* hbf = (__bf16*)alloc((size_t)TOTAL_NODES * DDIM * 2);
  __bf16* ebf = (__bf16*)alloc((size_t)TOTAL_EDGES * DDIM * 2);
  float* agg = (float*)alloc((size_t)TOTAL_NODES * MSGD * 4);
  __bf16* pW1m = (__bf16*)alloc((size_t)384 * 256 * 2);
  __bf16* pW2m = (__bf16*)alloc((size_t)256 * 256 * 2);
  __bf16* pW1r = (__bf16*)alloc((size_t)384 * 256 * 2);
  __bf16* pW2r = (__bf16*)alloc((size_t)256 * 256 * 2);
  __bf16* pU1 = (__bf16*)alloc((size_t)384 * 256 * 2);
  __bf16* pU2 = (__bf16*)alloc((size_t)256 * 128 * 2);
  float* tbuf = (float*)alloc((size_t)NUM_GRAPHS * 4096 * 4);

  pack_w_kernel<<<(384 * 256 + 255) / 256, 256, 0, stream>>>(msg_W1, pW1m, 384, 256);
  pack_w_kernel<<<(256 * 256 + 255) / 256, 256, 0, stream>>>(msg_W2, pW2m, 256, 256);
  pack_w_kernel<<<(384 * 256 + 255) / 256, 256, 0, stream>>>(rmsg_W1, pW1r, 384, 256);
  pack_w_kernel<<<(256 * 256 + 255) / 256, 256, 0, stream>>>(rmsg_W2, pW2r, 256, 256);
  pack_w_kernel<<<(384 * 256 + 255) / 256, 256, 0, stream>>>(upd_W1, pU1, 384, 256);
  pack_w_kernel<<<(256 * 128 + 255) / 256, 256, 0, stream>>>(upd_W2, pU2, 256, 128);

  enc_node_kernel<<<(TOTAL_NODES * DDIM) / 256, 256, 0, stream>>>(
      node_features, enc_node_W, enc_node_b, h, hbf);
  enc_edge_kernel<<<(TOTAL_EDGES * DDIM) / 256, 256, 0, stream>>>(
      edge_features, enc_edge_W, enc_edge_b, ebf);

  for (int step = 0; step < PROP_STEPS; step++) {
    hipMemsetAsync(agg, 0, (size_t)TOTAL_NODES * MSGD * 4, stream);
    msg_kernel<<<TOTAL_EDGES / 16, 128, 0, stream>>>(
        hbf, ebf, from_idx, to_idx, pW1m, pW2m, pW1r, pW2r,
        msg_b1, msg_b2, rmsg_b1, rmsg_b2, agg);
    upd_kernel<<<TOTAL_NODES / 16, 128, 0, stream>>>(
        agg, pU1, pU2, upd_b1, upd_b2, h, hbf);
  }

  sink_t_kernel<<<NUM_GRAPHS, 256, 0, stream>>>(h, sink_W1, sink_b1, sink_W2, sink_b2, tbuf);
  pair_kernel<<<NUM_GRAPHS / 2, 256, 0, stream>>>(h, tbuf, out);
}
